// EncodeProcessDecode_36945308680562
// MI455X (gfx1250) — matrix operand order repaired and checked by reading, not compiled, not run on hardware
//
#include <hip/hip_runtime.h>
#include <hip/hip_bf16.h>

typedef __bf16 bf16;
typedef __attribute__((ext_vector_type(16))) __bf16 v16bf;
typedef __attribute__((ext_vector_type(8)))  __bf16 v8bf;
typedef __attribute__((ext_vector_type(4)))  __bf16 v4bf;
typedef __attribute__((ext_vector_type(8)))  float  v8f;

static_assert(sizeof(bf16) == 2, "bf16 must be 2 bytes");

constexpr int LDIM = 128;   // hidden width
constexpr int NC   = 100000;
constexpr int NO   = 20000;
constexpr int EM   = 400000;
constexpr int EC   = 200000;
constexpr int EW   = 150000;
constexpr int NF   = 16;
constexpr int EFD  = 9;
constexpr int OUTD = 3;
constexpr int NSTEP= 2;

struct Seg { const float* ptr; const int* idx; int width; };

struct MlpArgs {
  Seg seg[4];
  int nrows;
  int din;
  int din_pad;     // din rounded up to 32
  int dout;        // valid output columns (128 or 3)
  int out_stride;
  int vec4;        // input staging can use float4 path
  const bf16 *WT0, *WT1, *WT2;    // pre-transposed bf16 weights [128 cols][Kpad]
  const float *b0, *b1, *b2, *g, *be;  // g/be null => no LN
  const float *res;                    // residual source or null (stride LDIM)
  float *out;
};

// ---------------------------------------------------------------------------
// Fused 3-layer MLP: out = [LN]( relu(relu(X@W0+b0)@W1+b1)@W2 + b2 ) [+res]
// Tile: 128 rows x 128 cols per 256-thread block.
// A fragments from LDS (gathered input / hidden activations, wave-local rows).
// B fragments straight from global bf16 weights (L2-resident, col-major):
// per lane one contiguous 32B read in the same contiguous K order as the A fragment.
// Single barrier in the whole kernel (after input staging).
// ---------------------------------------------------------------------------
__launch_bounds__(256)
__global__ void fused_mlp_kernel(MlpArgs a) {
  extern __shared__ char smem[];
  bf16* sX  = (bf16*)smem;                     // [128][din_pad]
  bf16* sH  = sX + (size_t)128 * a.din_pad;    // [128][128] hidden 0
  bf16* sH2 = sH + 128 * 128;                  // [128][128] hidden 1

  const int tid    = threadIdx.x;
  const int lane   = tid & 31;
  const int wave   = tid >> 5;
  const int hi     = lane >> 4;      // half-wave select
  const int ln16   = lane & 15;
  const int waveM0 = wave * 16;      // 8 waves x 16 rows = 128-row tile
  const int blockRow = blockIdx.x * 128;

  // ---- stage gathered/concatenated input tile as bf16 ----
  const int dinp = a.din_pad;
  if (a.vec4) {
    const int dinp4 = dinp >> 2;
    for (int t = tid; t < 128 * dinp4; t += 256) {
      int r = t / dinp4, k = (t - r * dinp4) << 2;
      float4 v = make_float4(0.f, 0.f, 0.f, 0.f);
      int gr = blockRow + r;
      if (gr < a.nrows && k < a.din) {
        int k2 = k, si = 0;
        while (k2 >= a.seg[si].width) { k2 -= a.seg[si].width; ++si; }
        Seg s = a.seg[si];
        if (s.ptr) {
          int sr = s.idx ? s.idx[gr] : gr;
          v = *(const float4*)&s.ptr[(size_t)sr * s.width + k2];
        }
      }
      v4bf o; o[0] = (bf16)v.x; o[1] = (bf16)v.y; o[2] = (bf16)v.z; o[3] = (bf16)v.w;
      *(v4bf*)&sX[(size_t)r * dinp + k] = o;
    }
  } else {
    for (int t = tid; t < 128 * dinp; t += 256) {
      int r = t / dinp, k = t - r * dinp;
      float v = 0.f;
      int gr = blockRow + r;
      if (gr < a.nrows && k < a.din) {
        int k2 = k, si = 0;
        while (k2 >= a.seg[si].width) { k2 -= a.seg[si].width; ++si; }
        Seg s = a.seg[si];
        if (s.ptr) {
          int sr = s.idx ? s.idx[gr] : gr;
          v = s.ptr[(size_t)sr * s.width + k2];
        }
      }
      sX[(size_t)r * dinp + k] = (bf16)v;
    }
  }
  __syncthreads();   // only barrier: staging is cross-wave, everything after is wave-local

  for (int layer = 0; layer < 3; ++layer) {
    const bf16* sA; int lda, kdim, Kpad;
    const bf16* Wt;
    const float* bias;
    bf16* sOut;
    if (layer == 0)      { sA = sX;  lda = dinp; kdim = dinp; Kpad = dinp;
                           Wt = a.WT0; bias = a.b0; sOut = sH;  }
    else if (layer == 1) { sA = sH;  lda = LDIM; kdim = LDIM; Kpad = LDIM;
                           Wt = a.WT1; bias = a.b1; sOut = sH2; }
    else                 { sA = sH2; lda = LDIM; kdim = LDIM; Kpad = LDIM;
                           Wt = a.WT2; bias = a.b2; sOut = nullptr; }

    v8f acc[8] = {};
    const int row = waveM0 + ln16;
    for (int kk = 0; kk < kdim; kk += 32) {
      // A fragment: K = kk + 16*hi .. +15 contiguous (matches B)
      v16bf afr = *(const v16bf*)&sA[(size_t)row * lda + kk + 16 * hi];

#pragma unroll
      for (int nt = 0; nt < 8; ++nt) {
        // B fragment: col = nt*16 + ln16, K = kk + 16*hi .. +15, contiguous 32B in WT
        int col = nt * 16 + ln16;
        v16bf bfr = *(const v16bf*)&Wt[(size_t)col * Kpad + kk + 16 * hi];
        acc[nt] = __builtin_amdgcn_wmma_f32_16x16x32_bf16(
            false, afr, false, bfr, (short)0, acc[nt], false, false);
      }
    }

    if (layer < 2) {
      // bias + relu -> hidden LDS (C/D layout: row = v + 8*hi, col = lane&15)
      // wave-local rows: no barrier needed (LDS in-order within wave)
#pragma unroll
      for (int nt = 0; nt < 8; ++nt) {
        int col = nt * 16 + ln16;
        float bv = bias[col];
#pragma unroll
        for (int v = 0; v < 8; ++v) {
          int r = waveM0 + v + 8 * hi;
          float y = acc[nt][v] + bv;
          sOut[(size_t)r * LDIM + col] = (bf16)fmaxf(y, 0.f);
        }
      }
    } else {
      // final epilogue: bias (+LN) (+residual) -> global
#pragma unroll
      for (int nt = 0; nt < 8; ++nt) {
        int col = nt * 16 + ln16;
        float bv = (col < a.dout) ? bias[col] : 0.f;
#pragma unroll
        for (int v = 0; v < 8; ++v) acc[nt][v] += bv;
      }
      if (a.g) {  // layernorm over 128 cols: reduce within 16-lane half
#pragma unroll
        for (int v = 0; v < 8; ++v) {
          float s = 0.f, s2 = 0.f;
#pragma unroll
          for (int nt = 0; nt < 8; ++nt) { float y = acc[nt][v]; s += y; s2 += y * y; }
#pragma unroll
          for (int m = 1; m < 16; m <<= 1) {
            s  += __shfl_xor(s,  m, 32);
            s2 += __shfl_xor(s2, m, 32);
          }
          float mean = s * (1.f / 128.f);
          float var  = s2 * (1.f / 128.f) - mean * mean;
          float rinv = rsqrtf(var + 1e-5f);
#pragma unroll
          for (int nt = 0; nt < 8; ++nt) {
            int col = nt * 16 + ln16;
            acc[nt][v] = (acc[nt][v] - mean) * rinv * a.g[col] + a.be[col];
          }
        }
      }
#pragma unroll
      for (int nt = 0; nt < 8; ++nt) {
        int col = nt * 16 + ln16;
        if (col >= a.dout) continue;
#pragma unroll
        for (int v = 0; v < 8; ++v) {
          int r = blockRow + waveM0 + v + 8 * hi;
          if (r < a.nrows) {
            float o = acc[nt][v];
            if (a.res) o += a.res[(size_t)r * LDIM + col];
            a.out[(size_t)r * a.out_stride + col] = o;
          }
        }
      }
    }
  }
}

// ---------------------------------------------------------------------------
// One-time weight transpose + fp32->bf16 convert: W[kdim][ncols] -> WT[128][Kpad]
__global__ void convert_w_kernel(const float* W, bf16* WT, int kdim, int ncols, int Kpad) {
  int i = blockIdx.x * 256 + threadIdx.x;
  if (i < 128 * Kpad) {
    int col = i / Kpad, k = i - col * Kpad;
    float v = (k < kdim && col < ncols) ? W[(size_t)k * ncols + col] : 0.f;
    WT[(size_t)col * Kpad + k] = (bf16)v;
  }
}

__global__ void zero_kernel(float* p, size_t n) {
  size_t i = (size_t)blockIdx.x * blockDim.x + threadIdx.x;
  if (i < n) p[i] = 0.f;
}

__global__ void axpy_kernel(float* dst, const float* src, size_t n) {
  size_t i = (size_t)blockIdx.x * blockDim.x + threadIdx.x;
  if (i < n) dst[i] += src[i];
}

__global__ void scatter_add_kernel(const float* u, const int* idx, float* acc, int nedges) {
  size_t i = (size_t)blockIdx.x * blockDim.x + threadIdx.x;
  if (i < (size_t)nedges * LDIM) {
    int e = (int)(i >> 7);
    int c = (int)(i & 127);
    atomicAdd(&acc[(size_t)idx[e] * LDIM + c], u[i]);
  }
}

// ---------------------------------------------------------------------------
static inline Seg mkseg(const float* p, int w, const int* idx = nullptr) { return Seg{p, idx, w}; }
static inline Seg zseg(int w) { return Seg{nullptr, nullptr, w}; }

struct WTset { bf16 *w0, *w1, *w2; };

static void launch_mlp(hipStream_t stream,
                       Seg s0, Seg s1, Seg s2, Seg s3, int nseg,
                       int nrows, int din, int vec4,
                       const WTset& wt,
                       const float* b0, const float* b1, const float* b2,
                       const float* g, const float* be,
                       const float* res, float* out, int dout, int out_stride) {
  MlpArgs a;
  a.seg[0] = s0; a.seg[1] = s1; a.seg[2] = s2; a.seg[3] = s3;
  for (int i = nseg; i < 4; ++i) a.seg[i] = Seg{nullptr, nullptr, 1 << 30};
  a.nrows = nrows; a.din = din; a.din_pad = (din + 31) & ~31;
  a.dout = dout; a.out_stride = out_stride; a.vec4 = vec4;
  a.WT0 = wt.w0; a.WT1 = wt.w1; a.WT2 = wt.w2;
  a.b0 = b0; a.b1 = b1; a.b2 = b2; a.g = g; a.be = be; a.res = res; a.out = out;
  size_t lds = (size_t)128 * a.din_pad * sizeof(bf16)
             + 2 * (size_t)128 * 128 * sizeof(bf16);
  dim3 grid((unsigned)((nrows + 127) / 128)), block(256);
  fused_mlp_kernel<<<grid, block, (unsigned)lds, stream>>>(a);
}

extern "C" void kernel_launch(void* const* d_in, const int* in_sizes, int n_in,
                              void* d_out, int out_size, void* d_ws, size_t ws_size,
                              hipStream_t stream) {
  (void)in_sizes; (void)n_in; (void)out_size; (void)ws_size;

  const float* x_cloth   = (const float*)d_in[0];
  const float* x_obst    = (const float*)d_in[1];
  const float* ef_mesh   = (const float*)d_in[2];
  const float* ef_coarse = (const float*)d_in[3];
  const float* ef_wd     = (const float*)d_in[4];
  const float* ef_wi     = (const float*)d_in[5];

  const float* EN[8]; for (int i = 0; i < 8; ++i) EN[i] = (const float*)d_in[6 + i];
  const float* EE[8]; for (int i = 0; i < 8; ++i) EE[i] = (const float*)d_in[14 + i];
  const float* PE[8]; for (int i = 0; i < 8; ++i) PE[i] = (const float*)d_in[22 + i];
  const float* PN[8]; for (int i = 0; i < 8; ++i) PN[i] = (const float*)d_in[30 + i];
  const float* DC[6]; for (int i = 0; i < 6; ++i) DC[i] = (const float*)d_in[38 + i];

  const int* ei_mesh   = (const int*)d_in[44];  // (2, EM)
  const int* ei_coarse = (const int*)d_in[45];  // (2, EC)
  const int* w_src     = (const int*)d_in[46];  // (EW,)
  const int* w_dst     = (const int*)d_in[47];  // (EW,)
  const int* eim0 = ei_mesh,   *eim1 = ei_mesh + EM;
  const int* eic0 = ei_coarse, *eic1 = ei_coarse + EC;

  // ---- workspace carve: activations (f32) then converted weights (bf16) ----
  float* ws = (float*)d_ws;
  size_t off = 0;
  auto alloc = [&](size_t n) { float* p = ws + off; off += n; return p; };
  float* v_c  = alloc((size_t)NC * LDIM);
  float* v_o  = alloc((size_t)NO * LDIM);
  float* e_m  = alloc((size_t)EM * LDIM);
  float* e_c  = alloc((size_t)EC * LDIM);
  float* e_wd = alloc((size_t)EW * LDIM);
  float* e_wi = alloc((size_t)EW * LDIM);
  float* u    = alloc((size_t)EM * LDIM);   // reused per edge family
  float* a_c  = alloc((size_t)NC * LDIM);
  float* a_m  = alloc((size_t)NC * LDIM);
  float* a_wd = alloc((size_t)NC * LDIM);
  float* a_wi = alloc((size_t)NO * LDIM);

  bf16* wsb = (bf16*)(ws + off);
  size_t boff = 0;
  auto balloc = [&](size_t n) { bf16* p = wsb + boff; boff += n; return p; };

  auto conv = [&](const float* W, bf16* WT, int kdim, int ncols, int Kpad) {
    int n = 128 * Kpad;
    convert_w_kernel<<<(unsigned)((n + 255) / 256), 256, 0, stream>>>(W, WT, kdim, ncols, Kpad);
  };
  auto make_wt = [&](const float* W0, const float* W1, const float* W2,
                     int din, int ncols2) -> WTset {
    int kp0 = (din + 31) & ~31;
    WTset t;
    t.w0 = balloc((size_t)128 * kp0);
    t.w1 = balloc((size_t)128 * LDIM);
    t.w2 = balloc((size_t)128 * LDIM);
    conv(W0, t.w0, din, LDIM, kp0);
    conv(W1, t.w1, LDIM, LDIM, LDIM);
    conv(W2, t.w2, LDIM, ncols2, LDIM);
    return t;
  };

  // pre-convert every weight matrix once (all weights ~1.8MB bf16: L2-resident)
  WTset wt_en = make_wt(EN[0], EN[2], EN[4], NF, LDIM);
  WTset wt_ee[3];
  for (int k = 0; k < 3; ++k)
    wt_ee[k] = make_wt(EE[0] + (size_t)k * EFD * LDIM,
                       EE[2] + (size_t)k * LDIM * LDIM,
                       EE[4] + (size_t)k * LDIM * LDIM, EFD, LDIM);
  WTset wt_pe[NSTEP][3];
  for (int s = 0; s < NSTEP; ++s)
    for (int k = 0; k < 3; ++k) {
      size_t i = (size_t)(s * 3 + k);
      wt_pe[s][k] = make_wt(PE[0] + i * (3 * LDIM) * LDIM,
                            PE[2] + i * LDIM * LDIM,
                            PE[4] + i * LDIM * LDIM, 3 * LDIM, LDIM);
    }
  WTset wt_pn[NSTEP];
  for (int s = 0; s < NSTEP; ++s)
    wt_pn[s] = make_wt(PN[0] + (size_t)s * (4 * LDIM) * LDIM,
                       PN[2] + (size_t)s * LDIM * LDIM,
                       PN[4] + (size_t)s * LDIM * LDIM, 4 * LDIM, LDIM);
  WTset wt_dc = make_wt(DC[0], DC[2], DC[4], LDIM, OUTD);

  auto zero = [&](float* p, size_t n) {
    zero_kernel<<<(unsigned)((n + 255) / 256), 256, 0, stream>>>(p, n);
  };
  auto axpy = [&](float* d, const float* s, size_t n) {
    axpy_kernel<<<(unsigned)((n + 255) / 256), 256, 0, stream>>>(d, s, n);
  };
  auto scat = [&](const float* uu, const int* idx, float* acc, int ne) {
    scatter_add_kernel<<<(unsigned)(((size_t)ne * LDIM + 255) / 256), 256, 0, stream>>>(uu, idx, acc, ne);
  };

  // ---- encoders ----
  launch_mlp(stream, mkseg(x_cloth, NF), zseg(1), zseg(1), zseg(1), 1, NC, NF, 1,
             wt_en, EN[1], EN[3], EN[5], EN[6], EN[7], nullptr, v_c, LDIM, LDIM);
  launch_mlp(stream, mkseg(x_obst, NF), zseg(1), zseg(1), zseg(1), 1, NO, NF, 1,
             wt_en, EN[1], EN[3], EN[5], EN[6], EN[7], nullptr, v_o, LDIM, LDIM);

  auto enc_edge = [&](const float* ef, int ne, int k, float* out) {
    size_t kB = (size_t)k * LDIM;
    launch_mlp(stream, mkseg(ef, EFD), zseg(1), zseg(1), zseg(1), 1, ne, EFD, 0,
               wt_ee[k], EE[1] + kB, EE[3] + kB, EE[5] + kB, EE[6] + kB, EE[7] + kB,
               nullptr, out, LDIM, LDIM);
  };
  enc_edge(ef_mesh,   EM, 0, e_m);
  enc_edge(ef_wd,     EW, 1, e_wd);
  enc_edge(ef_wi,     EW, 1, e_wi);
  enc_edge(ef_coarse, EC, 2, e_c);

  // ---- message passing steps ----
  for (int s = 0; s < NSTEP; ++s) {
    zero(a_c,  (size_t)NC * LDIM);
    zero(a_m,  (size_t)NC * LDIM);
    zero(a_wd, (size_t)NC * LDIM);
    zero(a_wi, (size_t)NO * LDIM);

    auto edge_mlp = [&](Seg s0, Seg s1, Seg s2, int ne, int k) {
      size_t i = (size_t)(s * 3 + k) * LDIM;
      launch_mlp(stream, s0, s1, s2, zseg(1), 3, ne, 3 * LDIM, 1,
                 wt_pe[s][k], PE[1] + i, PE[3] + i, PE[5] + i, PE[6] + i, PE[7] + i,
                 nullptr, u, LDIM, LDIM);
    };

    // coarse edges
    edge_mlp(mkseg(v_c, LDIM, eic0), mkseg(v_c, LDIM, eic1), mkseg(e_c, LDIM), EC, 0);
    scat(u, eic1, a_c, EC);
    axpy(e_c, u, (size_t)EC * LDIM);
    // mesh edges
    edge_mlp(mkseg(v_c, LDIM, eim0), mkseg(v_c, LDIM, eim1), mkseg(e_m, LDIM), EM, 1);
    scat(u, eim1, a_m, EM);
    axpy(e_m, u, (size_t)EM * LDIM);
    // world (directed) edges
    edge_mlp(mkseg(v_o, LDIM, w_src), mkseg(v_c, LDIM, w_dst), mkseg(e_wd, LDIM), EW, 2);
    scat(u, w_dst, a_wd, EW);
    axpy(e_wd, u, (size_t)EW * LDIM);
    // world (inverse) edges
    edge_mlp(mkseg(v_c, LDIM, w_dst), mkseg(v_o, LDIM, w_src), mkseg(e_wi, LDIM), EW, 2);
    scat(u, w_src, a_wi, EW);
    axpy(e_wi, u, (size_t)EW * LDIM);

    // node updates (residual, in-place; all segments row-aligned so safe)
    size_t sb = (size_t)s * LDIM;
    launch_mlp(stream, mkseg(v_c, LDIM), mkseg(a_c, LDIM), mkseg(a_m, LDIM), mkseg(a_wd, LDIM),
               4, NC, 4 * LDIM, 1, wt_pn[s],
               PN[1] + sb, PN[3] + sb, PN[5] + sb, PN[6] + sb, PN[7] + sb,
               v_c, v_c, LDIM, LDIM);
    launch_mlp(stream, mkseg(v_o, LDIM), zseg(LDIM), zseg(LDIM), mkseg(a_wi, LDIM),
               4, NO, 4 * LDIM, 1, wt_pn[s],
               PN[1] + sb, PN[3] + sb, PN[5] + sb, PN[6] + sb, PN[7] + sb,
               v_o, v_o, LDIM, LDIM);
  }

  // ---- decoder (no LN, dout = 3) ----
  launch_mlp(stream, mkseg(v_c, LDIM), zseg(1), zseg(1), zseg(1), 1, NC, LDIM, 1,
             wt_dc, DC[1], DC[3], DC[5], nullptr, nullptr,
             nullptr, (float*)d_out, OUTD, OUTD);
}
